// HANEncoder_15934328668887
// MI455X (gfx1250) — compile-verified, matrix operand order checked by reading, not run on hardware
//
#include <hip/hip_runtime.h>
#include <math.h>
#include <stdint.h>

typedef __attribute__((ext_vector_type(16))) __bf16 v16bf;
typedef __attribute__((ext_vector_type(8)))  __bf16 v8bf;
typedef __attribute__((ext_vector_type(8)))  float  v8f;
typedef __attribute__((ext_vector_type(4)))  float  v4f;

#define NN 50000
#define EE 300000

#define TM 64
#define TK 32

// ---------------------------------------------------------------------------
// weight pre-pass: WT[c*K+k] = bf16(W[k*C+c])
// ---------------------------------------------------------------------------
__global__ void wt_transpose(const float* __restrict__ W, __bf16* __restrict__ WT,
                             int K, int C)
{
    int i = blockIdx.x * blockDim.x + threadIdx.x;
    if (i >= K * C) return;
    int k = i / C, c = i - k * C;
    WT[(size_t)c * K + k] = (__bf16)W[i];
}

// ---------------------------------------------------------------------------
// WMMA GEMM: out = A[MxK] @ W[KxC] + bias  (bf16 multiply, f32 accumulate)
// A staged to LDS via global_load_async_to_lds_b128 (one DRAM read of A total:
// each block covers the FULL output width C = NT*32). B fragments are read
// directly from bf16 transposed weights WT[C][K] (L2/WGP$-resident).
// MODE 0: store result.  MODE 1: colsum[c] += sum_rows tanh(result)
// Block: 256 threads = 8 waves -> 64 x C tile; wave = 16 x (C/2),
// NT accumulators per wave, NT WMMAs per k-step.  K mult of 32.
// ---------------------------------------------------------------------------
template <int MODE, int NT>   // NT = C/32  (8 for C=256, 4 for C=128)
__global__ __launch_bounds__(256) void wmma_gemm(
    const float* __restrict__ Amat, const __bf16* __restrict__ WT,
    const float* __restrict__ bias, float* __restrict__ out,
    int M, int K, float* __restrict__ colsum)
{
    constexpr int C = NT * 32;
    __shared__ float sAf[TM][36];     // padded stride: bank spread + 16B align
    __shared__ float cred[C];

    const int t    = threadIdx.x;
    const int lane = t & 31;
    const int w    = t >> 5;
    const int wm   = w >> 1;          // 0..3
    const int wn   = w & 1;           // 0..1
    const int mBase = blockIdx.x * TM;

    const int kb   = (lane >> 4) * 8;          // K sub-block per half-wave
    const int mrow = wm * 16 + (lane & 15);
    const int ncb  = wn * (NT * 16) + (lane & 15);

    v8f acc[NT];
    #pragma unroll
    for (int j = 0; j < NT; ++j) acc[j] = (v8f){};

    for (int k0 = 0; k0 < K; k0 += TK) {
        // ---- A tile 64x32 f32: async copy global -> LDS (2 x 16B per thread)
        #pragma unroll
        for (int i = 0; i < 2; ++i) {
            int idx4 = t * 2 + i;                 // 0..511 float4 chunks
            int r    = idx4 >> 3;                 // 0..63
            int c4   = (idx4 & 7) * 4;            // 0..28
            int row  = mBase + r; if (row >= M) row = M - 1;  // clamp: OOB rows discarded later
            unsigned ldst = (unsigned)(uintptr_t)&sAf[r][c4];
            const float* gsrc = Amat + (size_t)row * K + k0 + c4;
            asm volatile("global_load_async_to_lds_b128 %0, %1, off"
                         :: "v"(ldst), "v"(gsrc) : "memory");
        }
        asm volatile("s_wait_asynccnt 0x0" ::: "memory");
        __syncthreads();

        // ---- A fragment 16x32 from LDS (4 x b128), convert f32->bf16
        const float* ar = &sAf[mrow][0];
        v4f a0 = *(const v4f*)(ar + kb);
        v4f a1 = *(const v4f*)(ar + kb + 4);
        v4f a2 = *(const v4f*)(ar + 16 + kb);
        v4f a3 = *(const v4f*)(ar + 16 + kb + 4);
        v16bf af;
        #pragma unroll
        for (int j = 0; j < 4; ++j) {
            af[j]      = (__bf16)a0[j];
            af[4 + j]  = (__bf16)a1[j];
            af[8 + j]  = (__bf16)a2[j];
            af[12 + j] = (__bf16)a3[j];
        }

        // ---- NT B fragments, NT WMMAs: A fragment reused across full width
        #pragma unroll
        for (int j = 0; j < NT; ++j) {
            const size_t brow = (size_t)(ncb + j * 16) * K + k0;
            const v8bf bA = *(const v8bf*)(WT + brow + kb);
            const v8bf bB = *(const v8bf*)(WT + brow + 16 + kb);
            v16bf bf = __builtin_shufflevector(bA, bB,
                        0,1,2,3,4,5,6,7,8,9,10,11,12,13,14,15);
            acc[j] = __builtin_amdgcn_wmma_f32_16x16x32_bf16(false, af, false, bf,
                                                             (short)0, acc[j], false, false);
        }
        __syncthreads();
    }

    // C/D layout: VGPR r -> row = r + 8*(lane>=16), col = lane&15
    const int rowOff = wm * 16 + ((lane >> 4) * 8);

    if (MODE == 0) {
        if (mBase + TM <= M) {                       // uniform fast path
            #pragma unroll
            for (int j = 0; j < NT; ++j) {
                int col = ncb + j * 16;
                float bv = bias[col];
                #pragma unroll
                for (int r = 0; r < 8; ++r) {
                    size_t row = (size_t)(mBase + rowOff + r);
                    out[row * C + col] = acc[j][r] + bv;
                }
            }
        } else {
            #pragma unroll
            for (int j = 0; j < NT; ++j) {
                int col = ncb + j * 16;
                float bv = bias[col];
                #pragma unroll
                for (int r = 0; r < 8; ++r) {
                    int row = mBase + rowOff + r;
                    if (row < M) out[(size_t)row * C + col] = acc[j][r] + bv;
                }
            }
        }
    } else {
        if (t < C) cred[t] = 0.f;
        __syncthreads();
        #pragma unroll
        for (int j = 0; j < NT; ++j) {
            int col = ncb + j * 16;
            float bv = bias[col];
            float l = 0.f;
            #pragma unroll
            for (int r = 0; r < 8; ++r) {
                int row = mBase + rowOff + r;
                if (row < M) l += tanhf(acc[j][r] + bv);
            }
            atomicAdd(&cred[col], l);
        }
        __syncthreads();
        if (t < C) atomicAdd(&colsum[t], cred[t]);
    }
}

// ---------------------------------------------------------------------------
// per-node attention scores: s[n,h] = sum_d xh[n, h*D+d] * a[h,d]
// ---------------------------------------------------------------------------
__global__ void scores_kernel(const float* __restrict__ X, const float* __restrict__ a,
                              float* __restrict__ s, int Nn, int H, int D)
{
    int i = blockIdx.x * blockDim.x + threadIdx.x;
    if (i >= Nn * H) return;
    int n = i / H, h = i - n * H;
    const float* xp = X + (size_t)n * H * D + (size_t)h * D;
    const float* ap = a + (size_t)h * D;
    float acc = 0.f;
    for (int d = 0; d < D; d += 4) {
        v4f x4 = *(const v4f*)(xp + d);
        v4f a4 = *(const v4f*)(ap + d);
        acc += x4[0]*a4[0] + x4[1]*a4[1] + x4[2]*a4[2] + x4[3]*a4[3];
    }
    s[i] = acc;
}

// monotonic float<->uint encoding for atomicMax on possibly-negative floats
__device__ __forceinline__ unsigned fenc(float f) {
    unsigned b = __float_as_uint(f);
    return (b & 0x80000000u) ? ~b : (b | 0x80000000u);
}
__device__ __forceinline__ float fdec(unsigned u) {
    unsigned b = (u & 0x80000000u) ? (u & 0x7FFFFFFFu) : ~u;
    return __uint_as_float(b);
}

__global__ void edge_pass1(const int* __restrict__ src, const int* __restrict__ dst,
                           const float* __restrict__ ss, const float* __restrict__ sd,
                           float* __restrict__ al, unsigned* __restrict__ amax, int H)
{
    int i = blockIdx.x * blockDim.x + threadIdx.x;
    if (i >= EE * H) return;
    int e = i / H, h = i - e * H;
    int s = src[e], d = dst[e];
    float v = ss[s * H + h] + sd[d * H + h];
    v = (v > 0.f) ? v : 0.2f * v;
    al[i] = v;
    atomicMax(&amax[d * H + h], fenc(v));
}

__global__ void edge_pass2(const int* __restrict__ dst, float* __restrict__ al,
                           const unsigned* __restrict__ amax, float* __restrict__ denom, int H)
{
    int i = blockIdx.x * blockDim.x + threadIdx.x;
    if (i >= EE * H) return;
    int e = i / H, h = i - e * H;
    int d = dst[e];
    float ex = expf(al[i] - fdec(amax[d * H + h]));
    al[i] = ex;
    atomicAdd(&denom[d * H + h], ex);
}

__global__ void edge_pass3(const int* __restrict__ src, const int* __restrict__ dst,
                           const float* __restrict__ ex, const float* __restrict__ denom,
                           const float* __restrict__ xh, float* __restrict__ agg,
                           int H, int D, int C)
{
    int chunks = C / 32;
    int i = blockIdx.x * blockDim.x + threadIdx.x;
    if (i >= EE * chunks) return;
    int e = i / chunks;
    int c0 = (i - e * chunks) * 32;
    int h = c0 / D;
    int s = src[e], d = dst[e];
    float alpha = ex[e * H + h] / (denom[d * H + h] + 1e-16f);
    const float* xs = xh + (size_t)s * C + c0;
    float* ag = agg + (size_t)d * C + c0;
    #pragma unroll
    for (int j = 0; j < 32; j += 4) {
        v4f v = *(const v4f*)(xs + j);
        atomicAdd(&ag[j],     v[0] * alpha);
        atomicAdd(&ag[j + 1], v[1] * alpha);
        atomicAdd(&ag[j + 2], v[2] * alpha);
        atomicAdd(&ag[j + 3], v[3] * alpha);
    }
}

__global__ void relu_inplace(float* __restrict__ x, size_t n)
{
    size_t i = (size_t)blockIdx.x * blockDim.x + threadIdx.x;
    if (i < n) x[i] = fmaxf(x[i], 0.f);
}

// semantic-attention weights for the paper group (metapaths writes=0, cites=2)
__global__ void sem_weights(const float* __restrict__ colsum, const float* __restrict__ q,
                            int C, float invN, float* __restrict__ wPaper)
{
    __shared__ float red[3][256];
    int t = threadIdx.x;
    for (int m = 0; m < 3; ++m)
        red[m][t] = (t < C) ? q[t] * colsum[m * C + t] * invN : 0.f;
    __syncthreads();
    for (int s = 128; s > 0; s >>= 1) {
        if (t < s) { red[0][t] += red[0][t+s]; red[1][t] += red[1][t+s]; red[2][t] += red[2][t+s]; }
        __syncthreads();
    }
    if (t == 0) {
        float s0 = red[0][0], s2 = red[2][0];
        float mx = fmaxf(s0, s2);
        float e0 = expf(s0 - mx), e2 = expf(s2 - mx);
        wPaper[0] = e0 / (e0 + e2);
        wPaper[1] = e2 / (e0 + e2);
    }
}

__global__ void combine2_k(const float* __restrict__ g0, const float* __restrict__ g1,
                           const float* __restrict__ w, float* __restrict__ out,
                           size_t n, int elu)
{
    size_t i = (size_t)blockIdx.x * blockDim.x + threadIdx.x;
    if (i >= n) return;
    float v = w[0] * g0[i] + w[1] * g1[i];
    out[i] = elu ? ((v > 0.f) ? v : expf(v) - 1.f) : v;
}

__global__ void combine1_k(const float* __restrict__ g0, float* __restrict__ out,
                           size_t n, int elu)
{
    size_t i = (size_t)blockIdx.x * blockDim.x + threadIdx.x;
    if (i >= n) return;
    float v = g0[i];
    out[i] = elu ? ((v > 0.f) ? v : expf(v) - 1.f) : v;
}

__global__ void l2norm_rows(float* __restrict__ x, int C)
{
    __shared__ float red[128];
    int row = blockIdx.x, t = threadIdx.x;
    float v = x[(size_t)row * C + t];
    red[t] = v * v;
    __syncthreads();
    for (int s = 64; s > 0; s >>= 1) {
        if (t < s) red[t] += red[t + s];
        __syncthreads();
    }
    float nrm = fmaxf(sqrtf(red[0]), 1e-12f);
    x[(size_t)row * C + t] = v / nrm;
}

// ---------------------------------------------------------------------------
static inline void gemm0(const float* A, const __bf16* WT, const float* b, float* out,
                         int M, int K, int C, hipStream_t s)
{
    dim3 g((M + TM - 1) / TM, 1);
    if (C == 256) wmma_gemm<0, 8><<<g, 256, 0, s>>>(A, WT, b, out, M, K, nullptr);
    else          wmma_gemm<0, 4><<<g, 256, 0, s>>>(A, WT, b, out, M, K, nullptr);
}
static inline void gemm1(const float* A, const __bf16* WT, const float* b, float* cs,
                         int M, int K, int C, hipStream_t s)
{
    dim3 g((M + TM - 1) / TM, 1);
    if (C == 256) wmma_gemm<1, 8><<<g, 256, 0, s>>>(A, WT, b, nullptr, M, K, cs);
    else          wmma_gemm<1, 4><<<g, 256, 0, s>>>(A, WT, b, nullptr, M, K, cs);
}
static inline int gdiv(size_t n, int b) { return (int)((n + b - 1) / b); }

extern "C" void kernel_launch(void* const* d_in, const int* in_sizes, int n_in,
                              void* d_out, int out_size, void* d_ws, size_t ws_size,
                              hipStream_t stream)
{
    (void)in_sizes; (void)n_in; (void)out_size; (void)ws_size;
    const int N = NN, E = EE;

    const float* x_paper  = (const float*)d_in[0];
    const float* x_author = (const float*)d_in[1];
    const int*   ei[3]    = { (const int*)d_in[2], (const int*)d_in[3], (const int*)d_in[4] };
    const float* W0p = (const float*)d_in[5];  const float* b0p = (const float*)d_in[6];
    const float* W0a = (const float*)d_in[7];  const float* b0a = (const float*)d_in[8];
    const float* a0[6]; for (int i = 0; i < 6; ++i) a0[i] = (const float*)d_in[9 + i];
    const float* Wk0 = (const float*)d_in[15]; const float* bk0 = (const float*)d_in[16];
    const float* q0  = (const float*)d_in[17];
    const float* W1p = (const float*)d_in[18]; const float* b1p = (const float*)d_in[19];
    const float* W1a = (const float*)d_in[20]; const float* b1a = (const float*)d_in[21];
    const float* a1[6]; for (int i = 0; i < 6; ++i) a1[i] = (const float*)d_in[22 + i];
    const float* Wk1 = (const float*)d_in[28]; const float* bk1 = (const float*)d_in[29];
    const float* q1  = (const float*)d_in[30];
    float* outp = (float*)d_out;

    // workspace layout (floats)
    float* ws = (float*)d_ws;
    size_t off = 0;
    float*    A  = ws + off; off += (size_t)N * 256;   // h0_paper -> combined paper (L0)
    float*    Bb = ws + off; off += (size_t)N * 256;   // h0_author -> combined author (L0)
    float*    S  = ws + off; off += (size_t)6 * N * 8; // node scores
    float*    AL = ws + off; off += (size_t)3 * E * 8; // edge logits / exp
    unsigned* MX = (unsigned*)(ws + off); off += (size_t)3 * N * 8;
    float*    DN = ws + off; off += (size_t)3 * N * 8;
    float*    G  = ws + off; off += (size_t)3 * N * 256; // L0 agg; L1: h1p,h1a,agg1[3]
    float*    CS = ws + off; off += 3 * 256;
    float*    WP = ws + off; off += 8;
    __bf16*   WT0p = (__bf16*)(ws + off);
    __bf16*   WT0a = WT0p + (size_t)768 * 256;
    __bf16*   WTk0 = WT0a + (size_t)768 * 256;
    __bf16*   WT1p = WTk0 + (size_t)256 * 256;
    __bf16*   WT1a = WT1p + (size_t)256 * 128;
    __bf16*   WTk1 = WT1a + (size_t)256 * 128;

    // ---- weight pre-pass: f32 -> bf16 transposed (L2-resident B operands)
    wt_transpose<<<gdiv((size_t)768*256, 256), 256, 0, stream>>>(W0p, WT0p, 768, 256);
    wt_transpose<<<gdiv((size_t)768*256, 256), 256, 0, stream>>>(W0a, WT0a, 768, 256);
    wt_transpose<<<gdiv((size_t)256*256, 256), 256, 0, stream>>>(Wk0, WTk0, 256, 256);
    wt_transpose<<<gdiv((size_t)256*128, 256), 256, 0, stream>>>(W1p, WT1p, 256, 128);
    wt_transpose<<<gdiv((size_t)256*128, 256), 256, 0, stream>>>(W1a, WT1a, 256, 128);
    wt_transpose<<<gdiv((size_t)128*128, 256), 256, 0, stream>>>(Wk1, WTk1, 128, 128);

    // ================= Layer 0 (C=256, H=8, D=32) =================
    gemm0(x_paper,  WT0p, b0p, A,  N, 768, 256, stream);
    gemm0(x_author, WT0a, b0a, Bb, N, 768, 256, stream);

    const size_t NH0 = (size_t)N * 8;
    // writes(author->paper), written_by(paper->author), cites(paper->paper)
    const float* srcX0[3] = { Bb, A, A };
    for (int r = 0; r < 3; ++r) {
        const float* dstX = (r == 1) ? Bb : A;
        scores_kernel<<<gdiv(NH0, 256), 256, 0, stream>>>(srcX0[r], a0[2*r],   S + (2*r)   * NH0, N, 8, 32);
        scores_kernel<<<gdiv(NH0, 256), 256, 0, stream>>>(dstX,     a0[2*r+1], S + (2*r+1) * NH0, N, 8, 32);
    }

    hipMemsetAsync(MX, 0, 3 * NH0 * 4, stream);
    hipMemsetAsync(DN, 0, 3 * NH0 * 4, stream);
    hipMemsetAsync(G,  0, (size_t)3 * N * 256 * 4, stream);

    const size_t EH0 = (size_t)E * 8;
    for (int r = 0; r < 3; ++r) {
        const int* src = ei[r]; const int* dst = ei[r] + E;
        edge_pass1<<<gdiv(EH0, 256), 256, 0, stream>>>(src, dst, S + (2*r)*NH0, S + (2*r+1)*NH0,
                                                       AL + r*EH0, MX + r*NH0, 8);
        edge_pass2<<<gdiv(EH0, 256), 256, 0, stream>>>(dst, AL + r*EH0, MX + r*NH0, DN + r*NH0, 8);
        edge_pass3<<<gdiv(EH0, 256), 256, 0, stream>>>(src, dst, AL + r*EH0, DN + r*NH0,
                                                       srcX0[r], G + (size_t)r*N*256, 8, 32, 256);
    }
    relu_inplace<<<gdiv((size_t)3 * N * 256, 256), 256, 0, stream>>>(G, (size_t)3 * N * 256);

    hipMemsetAsync(CS, 0, 3 * 256 * 4, stream);
    for (int m = 0; m < 3; ++m)
        gemm1(G + (size_t)m * N * 256, WTk0, bk0, CS + m * 256, N, 256, 256, stream);
    sem_weights<<<1, 256, 0, stream>>>(CS, q0, 256, 1.0f / N, WP);

    combine2_k<<<gdiv((size_t)N * 256, 256), 256, 0, stream>>>(G, G + (size_t)2*N*256, WP, A,
                                                               (size_t)N * 256, 1);
    combine1_k<<<gdiv((size_t)N * 256, 256), 256, 0, stream>>>(G + (size_t)N*256, Bb,
                                                               (size_t)N * 256, 1);

    // ================= Layer 1 (C=128, H=1, D=128) =================
    float* h1p  = G;
    float* h1a  = G + (size_t)N * 128;
    float* AGG1 = G + (size_t)2 * N * 128;

    gemm0(A,  WT1p, b1p, h1p, N, 256, 128, stream);
    gemm0(Bb, WT1a, b1a, h1a, N, 256, 128, stream);

    const size_t NH1 = (size_t)N;
    const float* srcX1[3] = { h1a, h1p, h1p };
    for (int r = 0; r < 3; ++r) {
        const float* dstX = (r == 1) ? h1a : h1p;
        scores_kernel<<<gdiv(NH1, 256), 256, 0, stream>>>(srcX1[r], a1[2*r],   S + (2*r)   * NH1, N, 1, 128);
        scores_kernel<<<gdiv(NH1, 256), 256, 0, stream>>>(dstX,     a1[2*r+1], S + (2*r+1) * NH1, N, 1, 128);
    }

    hipMemsetAsync(MX, 0, 3 * NH1 * 4, stream);
    hipMemsetAsync(DN, 0, 3 * NH1 * 4, stream);
    hipMemsetAsync(AGG1, 0, (size_t)3 * N * 128 * 4, stream);

    const size_t EH1 = (size_t)E;
    for (int r = 0; r < 3; ++r) {
        const int* src = ei[r]; const int* dst = ei[r] + E;
        edge_pass1<<<gdiv(EH1, 256), 256, 0, stream>>>(src, dst, S + (2*r)*NH1, S + (2*r+1)*NH1,
                                                       AL + r*EH1, MX + r*NH1, 1);
        edge_pass2<<<gdiv(EH1, 256), 256, 0, stream>>>(dst, AL + r*EH1, MX + r*NH1, DN + r*NH1, 1);
        edge_pass3<<<gdiv((size_t)E * 4, 256), 256, 0, stream>>>(src, dst, AL + r*EH1, DN + r*NH1,
                                                                 srcX1[r], AGG1 + (size_t)r*N*128,
                                                                 1, 128, 128);
    }
    relu_inplace<<<gdiv((size_t)3 * N * 128, 256), 256, 0, stream>>>(AGG1, (size_t)3 * N * 128);

    hipMemsetAsync(CS, 0, 3 * 128 * 4, stream);
    for (int m = 0; m < 3; ++m)
        gemm1(AGG1 + (size_t)m * N * 128, WTk1, bk1, CS + m * 128, N, 128, 128, stream);
    sem_weights<<<1, 256, 0, stream>>>(CS, q1, 128, 1.0f / N, WP + 2);

    combine2_k<<<gdiv((size_t)N * 128, 256), 256, 0, stream>>>(AGG1, AGG1 + (size_t)2*N*128, WP + 2,
                                                               outp, (size_t)N * 128, 0);
    combine1_k<<<gdiv((size_t)N * 128, 256), 256, 0, stream>>>(AGG1 + (size_t)N*128,
                                                               outp + (size_t)N*128,
                                                               (size_t)N * 128, 0);

    l2norm_rows<<<2 * N, 128, 0, stream>>>(outp, 128);
}
